// LSTMModel_51101520888101
// MI455X (gfx1250) — matrix operand order repaired and checked by reading, not compiled, not run on hardware
//
#include <hip/hip_runtime.h>

#define HDIM 64
#define GDIM 256     // 4*H
#define TLEN 512
#define BROWS 16     // batch rows per workgroup (one M-tile)
#define NTHREADS 128 // 4 waves (wave32)

typedef __attribute__((ext_vector_type(16))) _Float16 v16h;
typedef __attribute__((ext_vector_type(8)))  _Float16 v8h;
typedef __attribute__((ext_vector_type(8)))  float    v8f;

union V16H { v16h v; v8h h[2]; };

#define WMMA_F16(A, B, C) \
  __builtin_amdgcn_wmma_f32_16x16x32_f16(false, (A), false, (B), (short)0, (C), false, false)

// CDNA5 hardware v_tanh_f32 (TRANS op, co-executes with WMMA).
#if __has_builtin(__builtin_amdgcn_tanhf)
__device__ __forceinline__ float tanh_f(float x) { return __builtin_amdgcn_tanhf(x); }
#else
__device__ __forceinline__ float tanh_f(float x) {
  float e = __expf(-2.0f * x);
  return (1.0f - e) * __builtin_amdgcn_rcpf(1.0f + e);
}
#endif
// Gate weights/biases for i,f,o are PRE-SCALED by 0.5, so sigmoid(2*acc) needs no mul:
__device__ __forceinline__ float sig_pre(float y) {   // y = 0.5*gate_preact
  return fmaf(0.5f, tanh_f(y), 0.5f);
}

// A fragment (16x32 f16, M=lane&15): halves 0..15 = K kslab*32 + (lane>=16?16:0) .. +15
// (contiguous K, matching the B fragment order)
__device__ __forceinline__ v16h load_a_frag(const _Float16* hbuf, int lane, int kslab) {
  const int row = lane & 15;
  const int off = kslab * 32 + ((lane >= 16) ? 16 : 0);
  V16H u;
  u.h[0] = *(const v8h*)(hbuf + row * HDIM + off);
  u.h[1] = *(const v8h*)(hbuf + row * HDIM + off + 8);
  return u.v;
}

// B fragment from LDS f16 weights stored [g][k] (k contiguous): 16 contiguous K
// values at K = kslab*32 + (lane>=16?16:0), row g = gbase + lane&15.
__device__ __forceinline__ v16h load_b_frag_lds(const _Float16* wbuf, int lane,
                                                int gbase, int kslab) {
  const int g   = gbase + (lane & 15);
  const int off = kslab * 32 + ((lane >= 16) ? 16 : 0);
  return *(const v16h*)(wbuf + g * HDIM + off);
}

// Fragment loaded once from global f32, scaled, converted to f16 (register-resident).
__device__ __forceinline__ v16h load_b_frag_global(const float* __restrict__ w,
                                                   int lane, int gbase, int kslab,
                                                   float scale) {
  const int g   = gbase + (lane & 15);
  const int off = kslab * 32 + ((lane >= 16) ? 16 : 0);
  const float* p = w + g * HDIM + off;
  v16h r;
  #pragma unroll
  for (int i = 0; i < 16; ++i) r[i] = (_Float16)(scale * p[i]);
  return r;
}

__global__ __launch_bounds__(NTHREADS) void lstm2_fused(
    const float* __restrict__ x,
    const float* __restrict__ w_ih0, const float* __restrict__ w_hh0,
    const float* __restrict__ b_ih0, const float* __restrict__ b_hh0,
    const float* __restrict__ w_ih1, const float* __restrict__ w_hh1,
    const float* __restrict__ b_ih1, const float* __restrict__ b_hh1,
    const float* __restrict__ w_fc,  const float* __restrict__ b_fc,
    float* __restrict__ out)
{
  __shared__ float    s_xt[TLEN * BROWS];        // 32 KB, x transposed [t][row]
  __shared__ _Float16 s_w0[GDIM * HDIM];         // 32 KB, w_hh0 f16 (gate-scaled)
  __shared__ _Float16 s_w1x[GDIM * HDIM];        // 32 KB, w_ih1 f16 (gate-scaled)
  __shared__ _Float16 s_h0a[BROWS * HDIM], s_h0b[BROWS * HDIM];  // 2 KB each
  __shared__ _Float16 s_h1a[BROWS * HDIM], s_h1b[BROWS * HDIM];  // 2 KB each

  const int tid  = threadIdx.x;
  const int lane = tid & 31;
  const int wave = tid >> 5;              // 0..3: h-column slice [16w, 16w+16)
  const int row0 = blockIdx.x * BROWS;

  // ---- layer-1 recurrent weights: register-resident (8 x v16h = 64 VGPRs) ----
  v16h bw1h[8];                           // [gate*2 + kslab]
  float b0r[4], b1r[4], wi0r[4];
  #pragma unroll
  for (int g = 0; g < 4; ++g) {
    const float sc = (g == 2) ? 1.0f : 0.5f;   // fold sigmoid's 0.5 into i,f,o
    const int gb = g * HDIM + wave * 16;
    bw1h[g*2+0] = load_b_frag_global(w_hh1, lane, gb, 0, sc);
    bw1h[g*2+1] = load_b_frag_global(w_hh1, lane, gb, 1, sc);
    const int gc = gb + (lane & 15);
    b0r[g]  = sc * (b_ih0[gc] + b_hh0[gc]);
    b1r[g]  = sc * (b_ih1[gc] + b_hh1[gc]);
    wi0r[g] = sc * w_ih0[gc];
  }

  // ---- one-time staging: weights -> f16 LDS (gate-scaled); x tile; zero h ----
  for (int i = tid; i < GDIM * HDIM; i += NTHREADS) {
    const int gate = i >> 12;                 // 64*64 elements per gate block
    const float sc = (gate == 2) ? 1.0f : 0.5f;
    s_w0[i]  = (_Float16)(sc * w_hh0[i]);
    s_w1x[i] = (_Float16)(sc * w_ih1[i]);
  }
  for (int i = tid; i < TLEN * BROWS; i += NTHREADS) {
    const int r  = i >> 9;          // TLEN = 512
    const int tt = i & (TLEN - 1);  // consecutive tid -> consecutive t (coalesced)
    s_xt[tt * BROWS + r] = x[(size_t)(row0 + r) * TLEN + tt];
  }
  for (int i = tid; i < BROWS * HDIM; i += NTHREADS) {
    s_h0a[i] = (_Float16)0.0f;
    s_h1a[i] = (_Float16)0.0f;
  }
  __syncthreads();

  _Float16* h0r = s_h0a; _Float16* h0w = s_h0b;
  _Float16* h1r = s_h1a; _Float16* h1w = s_h1b;

  v8f c0, c1;
  #pragma unroll
  for (int j = 0; j < 8; ++j) { c0[j] = 0.0f; c1[j] = 0.0f; }

  for (int t = 0; t < TLEN; ++t) {
    // ---------- layer 0: gates = (x*w_ih0 + b) + h0 @ w_hh0^T ----------
    v16h a0 = load_a_frag(h0r, lane, 0);
    v16h a1 = load_a_frag(h0r, lane, 1);

    float xv[8];
    const float* xp = s_xt + t * BROWS + ((lane >= 16) ? 8 : 0);
    #pragma unroll
    for (int j = 0; j < 8; ++j) xv[j] = xp[j];

    v8f acc[4];
    #pragma unroll
    for (int g = 0; g < 4; ++g) {
      const int gb = g * HDIM + wave * 16;
      v8f a;
      #pragma unroll
      for (int j = 0; j < 8; ++j) a[j] = fmaf(xv[j], wi0r[g], b0r[g]);
      a = WMMA_F16(a0, load_b_frag_lds(s_w0, lane, gb, 0), a);
      a = WMMA_F16(a1, load_b_frag_lds(s_w0, lane, gb, 1), a);
      acc[g] = a;
    }

    float hv[8];
    #pragma unroll
    for (int j = 0; j < 8; ++j) {
      const float ig = sig_pre(acc[0][j]);
      const float fg = sig_pre(acc[1][j]);
      const float gg = tanh_f(acc[2][j]);
      const float og = sig_pre(acc[3][j]);
      const float c  = fmaf(fg, c0[j], ig * gg);
      c0[j] = c;
      hv[j] = og * tanh_f(c);
    }

    // write to the OTHER buffer -> no write-after-read barrier needed
    #pragma unroll
    for (int j = 0; j < 8; ++j) {
      const int r = (lane < 16) ? j : (j + 8);
      h0w[r * HDIM + wave * 16 + (lane & 15)] = (_Float16)hv[j];
    }
    __syncthreads();   // B1: h0_t visible to all waves

    // ---------- layer 1: gates = b1 + h0_t @ w_ih1^T + h1 @ w_hh1^T ----------
    v16h ax0 = load_a_frag(h0w, lane, 0);
    v16h ax1 = load_a_frag(h0w, lane, 1);
    v16h ah0 = load_a_frag(h1r, lane, 0);
    v16h ah1 = load_a_frag(h1r, lane, 1);

    #pragma unroll
    for (int g = 0; g < 4; ++g) {
      const int gb = g * HDIM + wave * 16;
      v8f a;
      #pragma unroll
      for (int j = 0; j < 8; ++j) a[j] = b1r[g];
      a = WMMA_F16(ax0, load_b_frag_lds(s_w1x, lane, gb, 0), a);
      a = WMMA_F16(ax1, load_b_frag_lds(s_w1x, lane, gb, 1), a);
      a = WMMA_F16(ah0, bw1h[g*2+0], a);
      a = WMMA_F16(ah1, bw1h[g*2+1], a);
      acc[g] = a;
    }

    #pragma unroll
    for (int j = 0; j < 8; ++j) {
      const float ig = sig_pre(acc[0][j]);
      const float fg = sig_pre(acc[1][j]);
      const float gg = tanh_f(acc[2][j]);
      const float og = sig_pre(acc[3][j]);
      const float c  = fmaf(fg, c1[j], ig * gg);
      c1[j] = c;
      hv[j] = og * tanh_f(c);
    }

    #pragma unroll
    for (int j = 0; j < 8; ++j) {
      const int r = (lane < 16) ? j : (j + 8);
      h1w[r * HDIM + wave * 16 + (lane & 15)] = (_Float16)hv[j];
    }
    __syncthreads();   // B2: h1_t visible for next iteration

    // swap double buffers
    _Float16* tmp;
    tmp = h0r; h0r = h0w; h0w = tmp;
    tmp = h1r; h1r = h1w; h1w = tmp;
  }

  // ---------- final FC: out[b] = h1_last . w_fc + b_fc ----------
  // last h1 was written to h1w then swapped into h1r; B2 already made it visible
  if (tid < BROWS) {
    float s = b_fc[0];
    #pragma unroll
    for (int k = 0; k < HDIM; ++k) s += (float)h1r[tid * HDIM + k] * w_fc[k];
    out[row0 + tid] = s;
  }
}

extern "C" void kernel_launch(void* const* d_in, const int* in_sizes, int n_in,
                              void* d_out, int out_size, void* d_ws, size_t ws_size,
                              hipStream_t stream) {
  const float* x     = (const float*)d_in[0];
  const float* w_ih0 = (const float*)d_in[1];
  const float* w_hh0 = (const float*)d_in[2];
  const float* b_ih0 = (const float*)d_in[3];
  const float* b_hh0 = (const float*)d_in[4];
  const float* w_ih1 = (const float*)d_in[5];
  const float* w_hh1 = (const float*)d_in[6];
  const float* b_ih1 = (const float*)d_in[7];
  const float* b_hh1 = (const float*)d_in[8];
  const float* w_fc  = (const float*)d_in[9];
  const float* b_fc  = (const float*)d_in[10];

  const int B = in_sizes[0] / TLEN;   // 2048
  dim3 grid(B / BROWS);               // 128 workgroups
  dim3 block(NTHREADS);
  lstm2_fused<<<grid, block, 0, stream>>>(x, w_ih0, w_hh0, b_ih0, b_hh0,
                                          w_ih1, w_hh1, b_ih1, b_hh1,
                                          w_fc, b_fc, (float*)d_out);
}
